// DSSConv2d_36438502540032
// MI455X (gfx1250) — matrix operand order repaired and checked by reading, not compiled, not run on hardware
//
#include <hip/hip_runtime.h>
#include <hip/hip_bf16.h>

// ---- types for WMMA intrinsics (gfx1250, wave32) ----
typedef __attribute__((ext_vector_type(16))) __bf16          v16bf;
typedef __attribute__((ext_vector_type(16))) unsigned short  v16u;
typedef unsigned short v16u_a16 __attribute__((ext_vector_type(16), aligned(16)));
typedef __attribute__((ext_vector_type(8)))  float           v8f;

#define RANKS   4
#define OC      128
#define ICH     128
#define BATCH   32
#define HH      56
#define WW      56
#define KTOT    (ICH * 9)           // 1152 reduction length
#define KSTEPS  (KTOT / 32)         // 36 bf16-WMMA k-chunks (tap-major: k' = tap*128 + ic)
#define MTILES  32                  // 512 / 16
#define SLAB_W  68                  // x = -1 .. 66 (covers padded n-tile 48..63 + 3x3 halo)
#define IC_STR  136                 // ic-dim stride in halves (68 words -> conflict-free b128)
#define SLAB_E  (3 * SLAB_W * IC_STR) // 27744 halves (~54 KB)
#define A_TILE_HALVES (KSTEPS * 512)  // halves between consecutive m-tiles in packed w

// fp32 -> bf16 round-to-nearest-even, as raw u16
static __device__ __forceinline__ unsigned short f2bf(float f) {
  unsigned int u = __float_as_uint(f);
  u += 0x7FFFu + ((u >> 16) & 1u);
  return (unsigned short)(u >> 16);
}

// -----------------------------------------------------------------------------
// Pre-pass: pack w[RANK,OC,IC,3,3] (fp32) into bf16 WMMA A-fragment order with
// tap-major K permutation:  k' = tap*128 + ic   (tap = ky*3+kx).
// Layout: [mtile(32)][kchunk(36)][lane(32)][slot(16)].
// A 16x32 bf16 k order used here (B matches): lane<16 holds row M=lane,   K = 0..15;
//                           lane>=16 holds row M=lane-16, K = 16..31.
// -----------------------------------------------------------------------------
__global__ void pack_w_kernel(const float* __restrict__ w, unsigned short* __restrict__ wp) {
  int idx = blockIdx.x * blockDim.x + threadIdx.x;
  if (idx >= MTILES * KSTEPS * 32 * 16) return;
  int slot = idx & 15;
  int lane = (idx >> 4) & 31;
  int kc   = (idx >> 9) % KSTEPS;
  int mt   = idx / (512 * KSTEPS);
  int m  = mt * 16 + (lane & 15);
  int kl = slot + (lane < 16 ? 0 : 16);
  int kp = kc * 32 + kl;            // permuted k'
  int tap = kp >> 7;                // 0..8
  int ic  = kp & 127;
  wp[idx] = f2bf(w[m * KTOT + ic * 9 + tap]);
}

// -----------------------------------------------------------------------------
// Main implicit-GEMM LRLC kernel. One workgroup per (batch b, output row y).
// 8 waves: wave = (chalf in {0,1}) x (ntile in {0..3});  N padded 56 -> 64.
// Each wave accumulates acc[rank][4 c-tiles] (16 tiles of v8f) over K=1152.
// LDS slab layout: [(ky*68 + col) * 136 + ic]  (bf16), col 0 <-> x = -1.
// Per k-step: B fragment = two ds_load_b128 per lane (32 consecutive bytes).
// -----------------------------------------------------------------------------
__global__ __launch_bounds__(256) void lrlc_wmma_kernel(
    const float* __restrict__ x,
    const unsigned short* __restrict__ wp,   // packed bf16 weights
    const float* __restrict__ cw_row,        // [RANK, H]
    const float* __restrict__ cw_col,        // [RANK, W]
    const float* __restrict__ b_row,         // [H]
    const float* __restrict__ b_col,         // [W]
    const float* __restrict__ bias_c,        // [OC]
    float* __restrict__ out)                 // [B, OC, H, W]
{
  __shared__ unsigned short sSlab[SLAB_E];

  const int b   = blockIdx.x / HH;
  const int y   = blockIdx.x % HH;
  const int tid = threadIdx.x;

  // ---- stage 3-row input slab (zero halo) into LDS as bf16, [ky][col][ic] ----
  for (int i = tid; i < ICH * 3 * SLAB_W; i += 256) {
    int col = i % SLAB_W;                 // x = col - 1
    int rk  = (i / SLAB_W) % 3;           // ky
    int ic  = i / (3 * SLAB_W);
    int xx  = col - 1;
    int yy  = y - 1 + rk;
    float v = 0.0f;
    if (xx >= 0 && xx < WW && yy >= 0 && yy < HH)
      v = x[((b * ICH + ic) * HH + yy) * WW + xx];
    sSlab[(rk * SLAB_W + col) * IC_STR + ic] = f2bf(v);
  }
  __syncthreads();

  const int wave  = tid >> 5;
  const int lane  = tid & 31;
  const int chalf = wave & 1;     // which 64-channel half
  const int nt    = wave >> 1;    // n-tile (16 output x's)
  const int nlane = lane & 15;    // column within n-tile
  const int lhalf = (lane < 16) ? 0 : 16;  // B: lane<16 -> K 0..15, lane>=16 -> K 16..31

  v8f acc[RANKS][4] = {};         // zero-init accumulators

#pragma unroll 1
  for (int kc = 0; kc < KSTEPS; ++kc) {
    // ---- wave-uniform k-chunk decode (SALU) ----
    const int tap = kc >> 2;            // 0..8
    const int ky  = tap / 3;
    const int kx  = tap - ky * 3;
    const int icb = (kc & 3) * 32;

    // ---- B fragment: 32 contiguous bytes -> 2x ds_load_b128, no repack ----
    const int col   = nt * 16 + nlane + kx;            // always < SLAB_W
    const int baddr = (ky * SLAB_W + col) * IC_STR + icb + lhalf;
    const v16u bu   = *(const v16u_a16*)(&sSlab[baddr]);
    const v16bf bfrag = __builtin_bit_cast(v16bf, bu);

    // ---- A fragments: one base pointer, compile-time tile offsets ----
    const unsigned short* abase =
        wp + ((size_t)((chalf * 4) * KSTEPS + kc) * 32 + lane) * 16;

    if (kc + 1 < KSTEPS)
      __builtin_prefetch(abase + 512, 0, 1);   // next k-chunk of first tile

    // ---- group 0: tiles i=0..7 (r=0..1): batch loads | barrier | batch WMMAs
    {
      v16u av[8];
#pragma unroll
      for (int i = 0; i < 8; ++i) {
        const int mtoff = (i >> 2) * 8 + (i & 3);      // r*8 + ct
        av[i] = *(const v16u*)(abase + (size_t)mtoff * A_TILE_HALVES);
      }
      __builtin_amdgcn_sched_barrier(0);   // keep all 8 fragments in flight
#pragma unroll
      for (int i = 0; i < 8; ++i) {
        acc[i >> 2][i & 3] = __builtin_amdgcn_wmma_f32_16x16x32_bf16(
            false, __builtin_bit_cast(v16bf, av[i]), false, bfrag,
            (short)0, acc[i >> 2][i & 3], false, false);
      }
    }
    // ---- group 1: tiles i=8..15 (r=2..3)
    {
      v16u av[8];
#pragma unroll
      for (int i = 0; i < 8; ++i) {
        const int mtoff = ((i + 8) >> 2) * 8 + (i & 3);
        av[i] = *(const v16u*)(abase + (size_t)mtoff * A_TILE_HALVES);
      }
      __builtin_amdgcn_sched_barrier(0);
#pragma unroll
      for (int i = 0; i < 8; ++i) {
        acc[2 + (i >> 2)][i & 3] = __builtin_amdgcn_wmma_f32_16x16x32_bf16(
            false, __builtin_bit_cast(v16bf, av[i]), false, bfrag,
            (short)0, acc[2 + (i >> 2)][i & 3], false, false);
      }
    }
  }

  // ---- epilogue: per-pixel softmax over rank, biases, store ----
  const int xq = nt * 16 + nlane;           // output x (may be >= 56 on last tile)
  const bool xvalid = (xq < WW);

  float lg[RANKS];
#pragma unroll
  for (int r = 0; r < RANKS; ++r)
    lg[r] = cw_row[r * HH + y] + (xvalid ? cw_col[r * WW + xq] : 0.0f);
  float mx = fmaxf(fmaxf(lg[0], lg[1]), fmaxf(lg[2], lg[3]));
  float sw[RANKS];
  float den = 0.0f;
#pragma unroll
  for (int r = 0; r < RANKS; ++r) { sw[r] = __expf(lg[r] - mx); den += sw[r]; }
  float inv = 1.0f / den;
#pragma unroll
  for (int r = 0; r < RANKS; ++r) sw[r] *= inv;

  const float brow = b_row[y];
  const float bcol = xvalid ? b_col[xq] : 0.0f;
  const int   joff = (lane < 16) ? 0 : 8;   // C/D layout: VGPR j -> row j or j+8

#pragma unroll
  for (int ct = 0; ct < 4; ++ct) {
#pragma unroll
    for (int j = 0; j < 8; ++j) {
      float v = acc[0][ct][j] * sw[0] + acc[1][ct][j] * sw[1] +
                acc[2][ct][j] * sw[2] + acc[3][ct][j] * sw[3];
      int c = chalf * 64 + ct * 16 + joff + j;
      v += bias_c[c] + brow + bcol;
      if (xvalid)
        out[((b * OC + c) * HH + y) * WW + xq] = v;
    }
  }
}

// -----------------------------------------------------------------------------
// Host launcher. Inputs (setup_inputs order):
//   0:x  1:w  2:cw_row  3:cw_col  4:b_row  5:b_col  6:b
// d_ws: packed bf16 weights (512*1152*2 = 1,179,648 bytes).
// -----------------------------------------------------------------------------
extern "C" void kernel_launch(void* const* d_in, const int* in_sizes, int n_in,
                              void* d_out, int out_size, void* d_ws, size_t ws_size,
                              hipStream_t stream) {
  (void)in_sizes; (void)n_in; (void)out_size; (void)ws_size;
  const float* x      = (const float*)d_in[0];
  const float* w      = (const float*)d_in[1];
  const float* cw_row = (const float*)d_in[2];
  const float* cw_col = (const float*)d_in[3];
  const float* b_rowp = (const float*)d_in[4];
  const float* b_colp = (const float*)d_in[5];
  const float* bias_c = (const float*)d_in[6];
  float* out = (float*)d_out;

  unsigned short* wp = (unsigned short*)d_ws;

  const int packN = MTILES * KSTEPS * 32 * 16;   // 589824
  pack_w_kernel<<<(packN + 255) / 256, 256, 0, stream>>>(w, wp);

  lrlc_wmma_kernel<<<BATCH * HH, 256, 0, stream>>>(
      x, wp, cw_row, cw_col, b_rowp, b_colp, bias_c, out);
}
